// EGraphSAGE_56057913147666
// MI455X (gfx1250) — matrix operand order repaired and checked by reading, not compiled, not run on hardware
//
#include <hip/hip_runtime.h>
#include <hip/hip_bf16.h>

// ---------------------------------------------------------------------------
// EGraphSAGE on MI455X (gfx1250): bf16 WMMA for the two big gathered GEMMs,
// f32 accumulation, float atomics for scatter-mean, VALU for the tiny
// 2-class edge predictor. Memory-bound workload (~3 GB moved, ~130us floor).
// ---------------------------------------------------------------------------

#define N_NODES 50000
#define N_EDGES 800000
#define D_NODE 128
#define D_EDGE 64
#define D_HID  128
#define KM     (D_NODE + D_EDGE)   // 192, message-GEMM K
#define KA     (2 * D_HID)         // 256, node-update K

typedef __attribute__((ext_vector_type(16))) __bf16          v16bf;
typedef __attribute__((ext_vector_type(16))) unsigned short  v16us;
typedef __attribute__((ext_vector_type(8)))  float           v8f;

__device__ __forceinline__ unsigned short f2bf(float f) {
  unsigned int u = __builtin_bit_cast(unsigned int, f);
  unsigned int r = u + 0x7FFFu + ((u >> 16) & 1u);   // round-to-nearest-even
  return (unsigned short)(r >> 16);
}

// --------------------------- small utility kernels -------------------------

__global__ void zero_f32_kernel(float* __restrict__ p, int n) {
  int i = blockIdx.x * blockDim.x + threadIdx.x;
  if (i < n) p[i] = 0.0f;
}

__global__ void cvt_bf16_kernel(const float* __restrict__ w,
                                unsigned short* __restrict__ o, int n) {
  int i = blockIdx.x * blockDim.x + threadIdx.x;
  if (i < n) o[i] = f2bf(w[i]);
}

__global__ void count_kernel(const int* __restrict__ dst,
                             float* __restrict__ cnt, int e) {
  int i = blockIdx.x * blockDim.x + threadIdx.x;
  if (i < e) atomicAdd(&cnt[dst[i]], 1.0f);
}

// ----------------- per-edge message GEMM + scatter-add ---------------------
// A = [h_src(128) | e(64)] per edge (16-edge tiles, bf16 in LDS),
// B = Wm (192x128 bf16, row major), wave w owns output cols [16w,16w+16).
// B fragments live in registers for the whole block (reused across tiles).

__global__ __launch_bounds__(256)
void msg_scatter_kernel(const float* __restrict__ hprev,      // [N,128]
                        const float* __restrict__ efeats,     // [E,64]
                        const int* __restrict__ src,
                        const int* __restrict__ dst,
                        const unsigned short* __restrict__ Wb, // [192,128] bf16
                        const float* __restrict__ bm,          // [128]
                        float* __restrict__ s,                 // [N,128] accum
                        int n_tiles)
{
  __shared__ unsigned short Atile[16 * KM];
  __shared__ int dtile[16];

  const int tid  = threadIdx.x;
  const int lane = tid & 31;
  const int wave = tid >> 5;
  const int half = lane >> 4;          // 0: lanes 0-15, 1: lanes 16-31
  const int l16  = lane & 15;
  const int col  = wave * 16 + l16;    // output column owned by this lane

  // Preload B fragments (K = 192 -> 6 chunks of 32). B layout: N = lane%16,
  // lanes 0-15 hold K=kb..kb+15, lanes 16-31 hold K=kb+16..kb+31.
  v16bf bfrag[6];
  #pragma unroll
  for (int kk = 0; kk < 6; ++kk) {
    v16us t;
    const int kb = kk * 32 + half * 16;
    #pragma unroll
    for (int i = 0; i < 16; ++i) t[i] = Wb[(kb + i) * D_HID + col];
    bfrag[kk] = __builtin_bit_cast(v16bf, t);
  }
  const float bias = bm[col];

  for (int tile = blockIdx.x; tile < n_tiles; tile += gridDim.x) {
    const int eb = tile * 16;
    // stage A tile (16 x 192) in LDS, converting fp32 -> bf16
    for (int idx = tid; idx < 16 * KM; idx += 256) {
      const int r = idx / KM, c = idx - r * KM;
      const int e = eb + r;
      const float v = (c < D_NODE) ? hprev[(long)src[e] * D_NODE + c]
                                   : efeats[(long)e * D_EDGE + (c - D_NODE)];
      Atile[idx] = f2bf(v);
    }
    if (tid < 16) dtile[tid] = dst[eb + tid];
    __syncthreads();

    v8f acc = {};
    #pragma unroll
    for (int kk = 0; kk < 6; ++kk) {
      // A layout: row = lane%16; lane half 0: K = kb..kb+15;
      //                          lane half 1: K = kb+16..kb+31.
      v16us t;
      const int b0 = kk * 32 + half * 16;
      #pragma unroll
      for (int i = 0; i < 8; ++i) t[i] = Atile[l16 * KM + b0 + i];
      #pragma unroll
      for (int i = 0; i < 8; ++i) t[8 + i] = Atile[l16 * KM + b0 + 8 + i];
      const v16bf af = __builtin_bit_cast(v16bf, t);
      acc = __builtin_amdgcn_wmma_f32_16x16x32_bf16(
          false, af, false, bfrag[kk], (short)0, acc, false, false);
    }

    // C/D layout: element r -> row (r + 8*half), col = wave*16 + lane%16
    #pragma unroll
    for (int r = 0; r < 8; ++r) {
      const int m = r + half * 8;
      atomicAdd(&s[(long)dtile[m] * D_HID + col], acc[r] + bias);
    }
    __syncthreads();
  }
}

// --------------------------- node update GEMM ------------------------------
// A = [h(128) | h_neigh(128)] per node (16-node tiles), B = Wa (256x128),
// fused bias + ReLU on store.

__global__ __launch_bounds__(256)
void node_update_kernel(const float* __restrict__ hprev,      // [N,128]
                        const float* __restrict__ s,          // [N,128]
                        const float* __restrict__ cnt,        // [N]
                        const unsigned short* __restrict__ Wb, // [256,128] bf16
                        const float* __restrict__ ba,          // [128]
                        float* __restrict__ hout,              // [N,128]
                        int n_tiles)
{
  __shared__ unsigned short Atile[16 * KA];
  __shared__ float invc[16];

  const int tid  = threadIdx.x;
  const int lane = tid & 31;
  const int wave = tid >> 5;
  const int half = lane >> 4;
  const int l16  = lane & 15;
  const int col  = wave * 16 + l16;

  v16bf bfrag[8];                      // K = 256 -> 8 chunks
  #pragma unroll
  for (int kk = 0; kk < 8; ++kk) {
    v16us t;
    const int kb = kk * 32 + half * 16;
    #pragma unroll
    for (int i = 0; i < 16; ++i) t[i] = Wb[(kb + i) * D_HID + col];
    bfrag[kk] = __builtin_bit_cast(v16bf, t);
  }
  const float bias = ba[col];

  for (int tile = blockIdx.x; tile < n_tiles; tile += gridDim.x) {
    const int nb = tile * 16;
    if (tid < 16) invc[tid] = 1.0f / fmaxf(cnt[nb + tid], 1.0f);
    __syncthreads();

    for (int idx = tid; idx < 16 * KA; idx += 256) {
      const int r = idx / KA, c = idx - r * KA;
      const long node = nb + r;
      const float v = (c < D_HID) ? hprev[node * D_HID + c]
                                  : s[node * D_HID + (c - D_HID)] * invc[r];
      Atile[idx] = f2bf(v);
    }
    __syncthreads();

    v8f acc = {};
    #pragma unroll
    for (int kk = 0; kk < 8; ++kk) {
      v16us t;
      const int b0 = kk * 32 + half * 16;
      #pragma unroll
      for (int i = 0; i < 8; ++i) t[i] = Atile[l16 * KA + b0 + i];
      #pragma unroll
      for (int i = 0; i < 8; ++i) t[8 + i] = Atile[l16 * KA + b0 + 8 + i];
      const v16bf af = __builtin_bit_cast(v16bf, t);
      acc = __builtin_amdgcn_wmma_f32_16x16x32_bf16(
          false, af, false, bfrag[kk], (short)0, acc, false, false);
    }

    #pragma unroll
    for (int r = 0; r < 8; ++r) {
      const int m = r + half * 8;
      hout[(long)(nb + m) * D_HID + col] = fmaxf(acc[r] + bias, 0.0f);
    }
    __syncthreads();
  }
}

// --------------------------- edge predictor --------------------------------
// score = [h_src | h_dst] @ Wp(256x2) + bp. N_CLS=2 -> WMMA would waste 7/8
// of the tile; bandwidth-bound VALU dot product instead.

__global__ void predict_kernel(const float* __restrict__ h,
                               const int* __restrict__ src,
                               const int* __restrict__ dst,
                               const float* __restrict__ Wp,  // [256,2]
                               const float* __restrict__ bp,  // [2]
                               float* __restrict__ out, int e)
{
  const int i = blockIdx.x * blockDim.x + threadIdx.x;
  if (i >= e) return;
  const float* hs = h + (long)src[i] * D_HID;
  const float* hd = h + (long)dst[i] * D_HID;
  float a0 = bp[0], a1 = bp[1];
  #pragma unroll 4
  for (int k = 0; k < D_HID; ++k) {
    const float v = hs[k];
    a0 += v * Wp[k * 2 + 0];
    a1 += v * Wp[k * 2 + 1];
  }
  #pragma unroll 4
  for (int k = 0; k < D_HID; ++k) {
    const float v = hd[k];
    a0 += v * Wp[(D_HID + k) * 2 + 0];
    a1 += v * Wp[(D_HID + k) * 2 + 1];
  }
  out[(long)i * 2 + 0] = a0;
  out[(long)i * 2 + 1] = a1;
}

// ------------------------------- launcher ----------------------------------

extern "C" void kernel_launch(void* const* d_in, const int* in_sizes, int n_in,
                              void* d_out, int out_size, void* d_ws, size_t ws_size,
                              hipStream_t stream) {
  (void)in_sizes; (void)n_in; (void)out_size; (void)ws_size;

  const float* nfeats = (const float*)d_in[0];
  const float* efeats = (const float*)d_in[1];
  const int*   src    = (const int*)d_in[2];
  const int*   dst    = (const int*)d_in[3];
  const float* Wm1 = (const float*)d_in[4];  const float* bm1 = (const float*)d_in[5];
  const float* Wa1 = (const float*)d_in[6];  const float* ba1 = (const float*)d_in[7];
  const float* Wm2 = (const float*)d_in[8];  const float* bm2 = (const float*)d_in[9];
  const float* Wa2 = (const float*)d_in[10]; const float* ba2 = (const float*)d_in[11];
  const float* Wp  = (const float*)d_in[12]; const float* bp  = (const float*)d_in[13];
  float* out = (float*)d_out;

  // workspace layout
  char* w = (char*)d_ws;
  float* s    = (float*)w;  w += (size_t)N_NODES * D_HID * sizeof(float);
  float* cnt  = (float*)w;  w += (size_t)N_NODES * sizeof(float);
  float* h1   = (float*)w;  w += (size_t)N_NODES * D_HID * sizeof(float);
  float* h2   = (float*)w;  w += (size_t)N_NODES * D_HID * sizeof(float);
  unsigned short* Wm1b = (unsigned short*)w; w += (size_t)KM * D_HID * 2;
  unsigned short* Wa1b = (unsigned short*)w; w += (size_t)KA * D_HID * 2;
  unsigned short* Wm2b = (unsigned short*)w; w += (size_t)KM * D_HID * 2;
  unsigned short* Wa2b = (unsigned short*)w; w += (size_t)KA * D_HID * 2;

  const int nm = KM * D_HID, na = KA * D_HID;
  cvt_bf16_kernel<<<(nm + 255) / 256, 256, 0, stream>>>(Wm1, Wm1b, nm);
  cvt_bf16_kernel<<<(na + 255) / 256, 256, 0, stream>>>(Wa1, Wa1b, na);
  cvt_bf16_kernel<<<(nm + 255) / 256, 256, 0, stream>>>(Wm2, Wm2b, nm);
  cvt_bf16_kernel<<<(na + 255) / 256, 256, 0, stream>>>(Wa2, Wa2b, na);

  // in-degree counts (shared by both layers)
  zero_f32_kernel<<<(N_NODES + 255) / 256, 256, 0, stream>>>(cnt, N_NODES);
  count_kernel<<<(N_EDGES + 255) / 256, 256, 0, stream>>>(dst, cnt, N_EDGES);

  const int sElems = N_NODES * D_HID;
  const int eTiles = N_EDGES / 16;   // 50000
  const int nTiles = N_NODES / 16;   // 3125

  // ---- layer 1
  zero_f32_kernel<<<(sElems + 255) / 256, 256, 0, stream>>>(s, sElems);
  msg_scatter_kernel<<<2048, 256, 0, stream>>>(nfeats, efeats, src, dst,
                                               Wm1b, bm1, s, eTiles);
  node_update_kernel<<<512, 256, 0, stream>>>(nfeats, s, cnt, Wa1b, ba1, h1, nTiles);

  // ---- layer 2
  zero_f32_kernel<<<(sElems + 255) / 256, 256, 0, stream>>>(s, sElems);
  msg_scatter_kernel<<<2048, 256, 0, stream>>>(h1, efeats, src, dst,
                                               Wm2b, bm2, s, eTiles);
  node_update_kernel<<<512, 256, 0, stream>>>(h1, s, cnt, Wa2b, ba2, h2, nTiles);

  // ---- edge predictor
  predict_kernel<<<(N_EDGES + 255) / 256, 256, 0, stream>>>(h2, src, dst, Wp, bp,
                                                            out, N_EDGES);
}
